// TemporalAttention_20985210208778
// MI455X (gfx1250) — matrix operand order repaired and checked by reading, not compiled, not run on hardware
//
#include <hip/hip_runtime.h>
#include <hip/hip_bf16.h>

// ---------------------------------------------------------------------------
// Problem constants (from reference)
// ---------------------------------------------------------------------------
#define EMBED 1024
#define HEADS 16
#define HDIM  64
#define SEQ   2048
#define BATCH 2
#define MTOT  (BATCH * SEQ)   // 4096 rows

typedef __attribute__((ext_vector_type(16))) _Float16 v16h;
typedef __attribute__((ext_vector_type(8)))  _Float16 v8h;
typedef __attribute__((ext_vector_type(8)))  float    v8f;

static __device__ inline v16h cat8(v8h a, v8h b) {
  return __builtin_shufflevector(a, b, 0,1,2,3,4,5,6,7,8,9,10,11,12,13,14,15);
}
static __device__ inline v16h ld32B(const _Float16* p) {
  v8h a = *(const v8h*)p;
  v8h b = *(const v8h*)(p + 8);
  return cat8(a, b);
}
static __device__ inline v8f wmma16(v16h a, v16h b, v8f c) {
  // D = A(16x32 f16) * B(32x16 f16) + C(16x16 f32)
  return __builtin_amdgcn_wmma_f32_16x16x32_f16(false, a, false, b, (short)0, c,
                                                false, false);
}

// ---------------------------------------------------------------------------
// K0a: transpose + convert a weight W[K][N] (f32, x@W layout) -> Wt[N][K] f16
// ---------------------------------------------------------------------------
__global__ void wt_transpose_kernel(const float* __restrict__ W,
                                    _Float16* __restrict__ Wt, int K, int N) {
  int i = blockIdx.x * 256 + threadIdx.x;   // i in [0, N*K)
  int n = i / K;
  int k = i - n * K;
  Wt[i] = (_Float16)W[(size_t)k * N + n];
}

// ---------------------------------------------------------------------------
// K0b: build concat(x, gathered temporal emb) in f16: xcat[4096][2048]
// ---------------------------------------------------------------------------
__global__ void build_xcat_kernel(const float* __restrict__ x,
                                  const int* __restrict__ tpos,
                                  const float* __restrict__ temb,
                                  _Float16* __restrict__ xcat) {
  int i = blockIdx.x * 256 + threadIdx.x;   // over 4096*2048
  int m = i >> 11;                          // row (b*SEQ + s)
  int c = i & 2047;                         // col in [0,2048)
  int b = m >> 11;
  int s = m & 2047;
  float v;
  if (c < EMBED) {
    v = x[((size_t)(b * SEQ + s)) * EMBED + c];
  } else {
    int p = tpos[s];
    v = temb[(size_t)p * EMBED + (c - EMBED)];
  }
  xcat[i] = (_Float16)v;
}

// ---------------------------------------------------------------------------
// K1: WMMA GEMM  out = (A[M][K] @ Bt[N][K]^T + bias) * scale
// 8 waves/block, 128x128 tile; wave w computes rows [m0, m0+16) x 128 cols.
// Double-buffered 8KB B tile in LDS; B-fragment ds loads software-pipelined
// against the WMMAs so s_wait_dscnt is partial, not 0.
// mode 0: f16 row-major [M][N]
// mode 1: f16 head-transposed store  vt[b][h][d][s]   (for V)
// mode 2: f32 row-major [M][N]                        (final output)
// ---------------------------------------------------------------------------
__global__ __launch_bounds__(256) void gemm_f16_kernel(
    const _Float16* __restrict__ A, const _Float16* __restrict__ Bt,
    const float* __restrict__ bias, void* __restrict__ outp,
    int M, int N, int K, float scale, int mode) {
  __shared__ _Float16 Bls[2][128 * 32];   // 2 x 8 KB B tile
  const int tid  = threadIdx.x;
  const int lane = tid & 31;
  const int wave = tid >> 5;
  const int lo   = lane & 15;
  const int hi   = lane >> 4;
  const int n0   = blockIdx.x * 128;
  const int m0   = blockIdx.y * 128 + wave * 16;

  const v8f vzero = {0.f, 0.f, 0.f, 0.f, 0.f, 0.f, 0.f, 0.f};
  v8f acc[8];
#pragma unroll
  for (int t = 0; t < 8; ++t) acc[t] = vzero;

  // A fragment source: row (m0+lo), chunks at +16*hi and +16*hi+8 per k-step
  const _Float16* arow = A + (size_t)(m0 + lo) * K + 16 * hi;
  // cooperative B tile loader: 256 threads x 32 bytes = 8 KB
  const int crow = tid >> 1;
  const int ccol = (tid & 1) * 16;
  const _Float16* csrc = Bt + (size_t)(n0 + crow) * K + ccol;
  const int cofs = crow * 32 + ccol;

  // Prologue: stage tile 0 into buffer 0.
  {
    v8h r0 = *(const v8h*)(csrc);
    v8h r1 = *(const v8h*)(csrc + 8);
    *(v8h*)(&Bls[0][cofs])     = r0;
    *(v8h*)(&Bls[0][cofs] + 8) = r1;
  }

  const int nk = K >> 5;
  for (int it = 0; it < nk; ++it) {
    const int k0  = it << 5;
    const int cur = it & 1;
    const bool hasNext = (it + 1) < nk;

    __syncthreads();   // buffer `cur` fully staged (compiler waits dscnt first)

    // A fragment for this k-step (issued before next-tile loads so the first
    // WMMA only needs a partial loadcnt wait).
    v8h a0 = *(const v8h*)(arow + k0);
    v8h a1 = *(const v8h*)(arow + k0 + 8);
    if (hasNext) __builtin_prefetch((const void*)(arow + k0 + 32), 0, 1);

    // Fetch next B tile into registers (overlaps with the 8 WMMAs below).
    v8h nr0, nr1;
    if (hasNext) {
      nr0 = *(const v8h*)(csrc + k0 + 32);
      nr1 = *(const v8h*)(csrc + k0 + 40);
    }

    v16h a = cat8(a0, a1);
    const _Float16* base = Bls[cur];

    // Software-pipelined B-fragment feed from LDS.
    const _Float16* bp0 = base + (0 * 16 + lo) * 32 + 16 * hi;
    v16h bcurf = cat8(*(const v8h*)bp0, *(const v8h*)(bp0 + 8));
#pragma unroll
    for (int t = 0; t < 8; ++t) {
      v16h bnext = bcurf;
      if (t < 7) {
        const _Float16* bp = base + ((t + 1) * 16 + lo) * 32 + 16 * hi;
        bnext = cat8(*(const v8h*)bp, *(const v8h*)(bp + 8));
      }
      acc[t] = wmma16(a, bcurf, acc[t]);
      bcurf = bnext;
    }

    // Stage next tile into the other buffer (visible after next barrier).
    if (hasNext) {
      _Float16* d = &Bls[cur ^ 1][cofs];
      *(v8h*)(d)     = nr0;
      *(v8h*)(d + 8) = nr1;
    }
  }

#pragma unroll
  for (int t = 0; t < 8; ++t) {
    const int n = n0 + t * 16 + lo;
    const float bv = bias[n];
#pragma unroll
    for (int r = 0; r < 8; ++r) {
      const int m = m0 + r + 8 * hi;
      const float val = (acc[t][r] + bv) * scale;
      if (mode == 0) {
        ((_Float16*)outp)[(size_t)m * N + n] = (_Float16)val;
      } else if (mode == 2) {
        ((float*)outp)[(size_t)m * N + n] = val;
      } else {  // mode 1: vt[((b*16+h)*64+d)][s]
        const int b = m >> 11, s = m & 2047;
        const int h = n >> 6,  d = n & 63;
        ((_Float16*)outp)[((size_t)((b * HEADS + h) * HDIM + d)) * SEQ + s] =
            (_Float16)val;
      }
    }
  }
}

// ---------------------------------------------------------------------------
// K2: flash attention. Grid (B*H, SEQ/128), block 256 (8 waves x 16 q-rows).
// q pre-scaled by 1/sqrt(hd). k row-major [m][1024]; vt head-transposed.
// Next-step K fragments are issued right after the score WMMAs and V
// fragments after the softmax, so global latency hides under exp/shfl/LDS.
// ---------------------------------------------------------------------------
__global__ __launch_bounds__(256) void attn_kernel(
    const _Float16* __restrict__ q, const _Float16* __restrict__ k,
    const _Float16* __restrict__ vt, _Float16* __restrict__ att) {
  __shared__ _Float16 Plds[8][16 * 32];
  const int lane = threadIdx.x & 31;
  const int wave = threadIdx.x >> 5;
  const int lo   = lane & 15;
  const int hi   = lane >> 4;
  const int bh = blockIdx.x;
  const int b  = bh >> 4;
  const int h  = bh & 15;
  const int s0q = blockIdx.y * 128 + wave * 16;  // first query row of this wave

  // Q fragments (resident across whole key loop): dims [0,32) and [32,64)
  const _Float16* qrow =
      q + (size_t)(b * SEQ + s0q + lo) * EMBED + h * HDIM + 16 * hi;
  v16h qa0 = cat8(*(const v8h*)(qrow),      *(const v8h*)(qrow + 8));
  v16h qa1 = cat8(*(const v8h*)(qrow + 32), *(const v8h*)(qrow + 40));

  const _Float16* kbase = k + (size_t)(b * SEQ + lo) * EMBED + h * HDIM + 16 * hi;
  const _Float16* vbase = vt + (size_t)bh * HDIM * SEQ;
  _Float16* P = Plds[wave];

  const v8f vzero = {0.f, 0.f, 0.f, 0.f, 0.f, 0.f, 0.f, 0.f};
  v8f O[4];
#pragma unroll
  for (int t = 0; t < 4; ++t) O[t] = vzero;
  float mrun[8], lrun[8];
#pragma unroll
  for (int r = 0; r < 8; ++r) { mrun[r] = -1e30f; lrun[r] = 0.f; }

  // Preload K fragments for the first 32-key step.
  v16h kb00 = ld32B(kbase);
  v16h kb01 = ld32B(kbase + 32);
  v16h kb10 = ld32B(kbase + (size_t)16 * EMBED);
  v16h kb11 = ld32B(kbase + (size_t)16 * EMBED + 32);

  for (int n0 = 0; n0 < SEQ; n0 += 32) {
    // --- scores for the current 32 keys ---
    v8f sc0 = vzero, sc1 = vzero;
    sc0 = wmma16(qa0, kb00, sc0);
    sc0 = wmma16(qa1, kb01, sc0);
    sc1 = wmma16(qa0, kb10, sc1);
    sc1 = wmma16(qa1, kb11, sc1);

    // --- issue next step's K fragments (hidden under softmax below) ---
    const bool hasNext = (n0 + 32) < SEQ;
    v16h nk00 = kb00, nk01 = kb01, nk10 = kb10, nk11 = kb11;
    if (hasNext) {
      const _Float16* kr0 = kbase + (size_t)(n0 + 32) * EMBED;
      const _Float16* kr1 = kr0 + (size_t)16 * EMBED;
      nk00 = ld32B(kr0);
      nk01 = ld32B(kr0 + 32);
      nk10 = ld32B(kr1);
      nk11 = ld32B(kr1 + 32);
    }

    // --- online softmax over the 16 rows held by this wave ---
    float alpha[8];
#pragma unroll
    for (int r = 0; r < 8; ++r) {
      float mx = fmaxf(sc0[r], sc1[r]);
      mx = fmaxf(mx, __shfl_xor(mx, 1, 32));
      mx = fmaxf(mx, __shfl_xor(mx, 2, 32));
      mx = fmaxf(mx, __shfl_xor(mx, 4, 32));
      mx = fmaxf(mx, __shfl_xor(mx, 8, 32));
      const float mnew = fmaxf(mrun[r], mx);
      const float al = __expf(mrun[r] - mnew);
      const float p0 = __expf(sc0[r] - mnew);
      const float p1 = __expf(sc1[r] - mnew);
      const int row = r + 8 * hi;
      P[row * 32 + lo]      = (_Float16)p0;
      P[row * 32 + 16 + lo] = (_Float16)p1;
      float sum = p0 + p1;
      sum += __shfl_xor(sum, 1, 32);
      sum += __shfl_xor(sum, 2, 32);
      sum += __shfl_xor(sum, 4, 32);
      sum += __shfl_xor(sum, 8, 32);
      lrun[r] = lrun[r] * al + sum;
      mrun[r] = mnew;
      alpha[r] = al;
    }

    // --- issue V fragments (independent of the LDS P round-trip) ---
    v16h vb[4];
#pragma unroll
    for (int t = 0; t < 4; ++t) {
      const _Float16* vr = vbase + (size_t)(t * 16 + lo) * SEQ + n0 + 16 * hi;
      vb[t] = ld32B(vr);
    }

#pragma unroll
    for (int t = 0; t < 4; ++t)
#pragma unroll
      for (int r = 0; r < 8; ++r) O[t][r] *= alpha[r];

    // Re-stripe P (C layout) -> A fragment via per-wave LDS (in-order DS).
    const _Float16* prow = P + lo * 32 + 16 * hi;
    v16h pa = cat8(*(const v8h*)(prow), *(const v8h*)(prow + 8));

#pragma unroll
    for (int t = 0; t < 4; ++t) O[t] = wmma16(pa, vb[t], O[t]);

    kb00 = nk00; kb01 = nk01; kb10 = nk10; kb11 = nk11;
  }

#pragma unroll
  for (int r = 0; r < 8; ++r) {
    const float inv = 1.0f / lrun[r];
    const int row = r + 8 * hi;
#pragma unroll
    for (int t = 0; t < 4; ++t) {
      att[(size_t)(b * SEQ + s0q + row) * EMBED + h * HDIM + t * 16 + lo] =
          (_Float16)(O[t][r] * inv);
    }
  }
}

// ---------------------------------------------------------------------------
// Launch
// ---------------------------------------------------------------------------
extern "C" void kernel_launch(void* const* d_in, const int* in_sizes, int n_in,
                              void* d_out, int out_size, void* d_ws,
                              size_t ws_size, hipStream_t stream) {
  (void)in_sizes; (void)n_in; (void)out_size; (void)ws_size;
  const float* x    = (const float*)d_in[0];
  const int*   tpos = (const int*)d_in[1];
  const float* temb = (const float*)d_in[2];
  const float* Wmix = (const float*)d_in[3];
  const float* bmix = (const float*)d_in[4];
  const float* Wq   = (const float*)d_in[5];
  const float* bq   = (const float*)d_in[6];
  const float* Wk   = (const float*)d_in[7];
  const float* bk   = (const float*)d_in[8];
  const float* Wv   = (const float*)d_in[9];
  const float* bv   = (const float*)d_in[10];
  const float* Wo   = (const float*)d_in[11];
  const float* bo   = (const float*)d_in[12];

  char* ws = (char*)d_ws;
  _Float16* xcat = (_Float16*)ws; ws += (size_t)MTOT * 2 * EMBED * 2;   // 16 MB
  _Float16* WtMix = (_Float16*)ws; ws += (size_t)EMBED * 2 * EMBED * 2; //  4 MB
  _Float16* WtQ = (_Float16*)ws;  ws += (size_t)EMBED * EMBED * 2;      //  2 MB
  _Float16* WtK = (_Float16*)ws;  ws += (size_t)EMBED * EMBED * 2;
  _Float16* WtV = (_Float16*)ws;  ws += (size_t)EMBED * EMBED * 2;
  _Float16* WtO = (_Float16*)ws;  ws += (size_t)EMBED * EMBED * 2;
  _Float16* xt  = (_Float16*)ws;  ws += (size_t)MTOT * EMBED * 2;       //  8 MB
  _Float16* qb  = (_Float16*)ws;  ws += (size_t)MTOT * EMBED * 2;
  _Float16* kb  = (_Float16*)ws;  ws += (size_t)MTOT * EMBED * 2;
  _Float16* vtb = (_Float16*)ws;  ws += (size_t)MTOT * EMBED * 2;
  _Float16* atb = (_Float16*)ws;  ws += (size_t)MTOT * EMBED * 2;

  // --- weight transposes (f32 [K][N] -> f16 [N][K]) ---
  wt_transpose_kernel<<<(EMBED * 2 * EMBED) / 256, 256, 0, stream>>>(
      Wmix, WtMix, 2 * EMBED, EMBED);
  wt_transpose_kernel<<<(EMBED * EMBED) / 256, 256, 0, stream>>>(Wq, WtQ, EMBED, EMBED);
  wt_transpose_kernel<<<(EMBED * EMBED) / 256, 256, 0, stream>>>(Wk, WtK, EMBED, EMBED);
  wt_transpose_kernel<<<(EMBED * EMBED) / 256, 256, 0, stream>>>(Wv, WtV, EMBED, EMBED);
  wt_transpose_kernel<<<(EMBED * EMBED) / 256, 256, 0, stream>>>(Wo, WtO, EMBED, EMBED);

  // --- concat(x, temb) in f16 ---
  build_xcat_kernel<<<((size_t)MTOT * 2 * EMBED) / 256, 256, 0, stream>>>(
      x, tpos, temb, xcat);

  // --- mix GEMM: xt = xcat @ Wmix + bmix ---
  {
    dim3 g(EMBED / 128, MTOT / 128);
    gemm_f16_kernel<<<g, 256, 0, stream>>>(xcat, WtMix, bmix, xt, MTOT, EMBED,
                                           2 * EMBED, 1.0f, 0);
  }
  // --- Q/K/V projections (Q pre-scaled by 1/sqrt(hd)=0.125; V head-transposed) ---
  {
    dim3 g(EMBED / 128, MTOT / 128);
    gemm_f16_kernel<<<g, 256, 0, stream>>>(xt, WtQ, bq, qb, MTOT, EMBED, EMBED,
                                           0.125f, 0);
    gemm_f16_kernel<<<g, 256, 0, stream>>>(xt, WtK, bk, kb, MTOT, EMBED, EMBED,
                                           1.0f, 0);
    gemm_f16_kernel<<<g, 256, 0, stream>>>(xt, WtV, bv, vtb, MTOT, EMBED, EMBED,
                                           1.0f, 1);
  }
  // --- attention ---
  {
    dim3 g(BATCH * HEADS, SEQ / 128);
    attn_kernel<<<g, 256, 0, stream>>>(qb, kb, vtb, atb);
  }
  // --- output projection -> fp32 d_out ---
  {
    dim3 g(EMBED / 128, MTOT / 128);
    gemm_f16_kernel<<<g, 256, 0, stream>>>(atb, WtO, bo, d_out, MTOT, EMBED,
                                           EMBED, 1.0f, 2);
  }
}
